// SemiCausalAttention_11184094839359
// MI455X (gfx1250) — compile-verified, matrix operand order checked by reading, not run on hardware
//
#include <hip/hip_runtime.h>
#include <hip/hip_bf16.h>
#include <math.h>

// ---- problem constants (match reference) ----
#define BB     2
#define TT     2048
#define DM     1024
#define NH     16
#define DK     64
#define BT     (BB * TT)          // 4096 rows
#define N3     (3 * DM)           // 3072
#define SCALE  0.125f             // 1/sqrt(64)
#define LN_EPS 1e-6f
#define NORM_EPS 1e-8f
#define NEGV   (-10000.0f)

typedef __attribute__((ext_vector_type(16))) _Float16 v16h;
typedef __attribute__((ext_vector_type(8)))  _Float16 v8h;
typedef __attribute__((ext_vector_type(4)))  _Float16 v4h;
typedef __attribute__((ext_vector_type(8)))  float    v8f;
typedef __attribute__((ext_vector_type(4)))  float    v4f;
typedef __attribute__((ext_vector_type(4)))  int      v4i;

// Address-space-typed pointers for the async copy builtin (param types are
// v4i* in global (AS1) and LDS (AS3) address spaces, per hipcc diagnostic).
typedef __attribute__((address_space(1))) v4i as1_v4i;
typedef __attribute__((address_space(3))) v4i as3_v4i;

#if defined(__has_builtin)
#  if __has_builtin(__builtin_amdgcn_global_load_async_to_lds_b128)
#    define HAVE_ASYNC_LDS 1
#  endif
#  if __has_builtin(__builtin_amdgcn_s_wait_asynccnt)
#    define HAVE_WAIT_ASYNC 1
#  endif
#endif

__device__ __forceinline__ void wait_async_copies() {
#if defined(HAVE_ASYNC_LDS)
#  if defined(HAVE_WAIT_ASYNC)
  __builtin_amdgcn_s_wait_asynccnt(0);
#  else
  asm volatile("s_wait_asynccnt 0x0" ::: "memory");
#  endif
#endif
}

// Load a 16-bit WMMA A/B operand from a row-major row pointer.
// ISA 7.12.2: lanes 0-15 hold K {k0+0..7, k0+16..23}; lanes 16-31 hold
// K {k0+8..15, k0+24..31}. hi = (lane >= 16).
__device__ __forceinline__ v16h load_op16(const _Float16* __restrict__ row, int k0, int hi) {
  const int o0 = k0 + (hi ? 8 : 0);
  const int o1 = k0 + (hi ? 24 : 16);
  v8h lo = *reinterpret_cast<const v8h*>(row + o0);
  v8h hh = *reinterpret_cast<const v8h*>(row + o1);
  v16h r;
#pragma unroll
  for (int i = 0; i < 8; ++i) { r[i] = lo[i]; r[8 + i] = hh[i]; }
  return r;
}

// Same operand gather but from an LDS tile with padded row stride (in halves).
#define LDS_STRIDE 40
__device__ __forceinline__ v16h lds_op16(const _Float16* tile, int row, int hi) {
  const int o0 = hi ? 8 : 0;
  const int o1 = hi ? 24 : 16;
  v8h lo = *reinterpret_cast<const v8h*>(tile + row * LDS_STRIDE + o0);
  v8h hh = *reinterpret_cast<const v8h*>(tile + row * LDS_STRIDE + o1);
  v16h r;
#pragma unroll
  for (int i = 0; i < 8; ++i) { r[i] = lo[i]; r[8 + i] = hh[i]; }
  return r;
}

// ---------------- 1) LayerNorm -> f16 ----------------
__global__ void ln_kernel(const float* __restrict__ x, const float* __restrict__ gamma,
                          const float* __restrict__ beta, _Float16* __restrict__ xh) {
  const int row = blockIdx.x;             // 0..BT-1
  const int tid = threadIdx.x;            // 256 threads, 4 elems each
  const float* rp = x + (size_t)row * DM;
  v4f v = *reinterpret_cast<const v4f*>(rp + tid * 4);
  float s = v[0] + v[1] + v[2] + v[3];
  float ss = v[0] * v[0] + v[1] * v[1] + v[2] * v[2] + v[3] * v[3];
  __shared__ float r1[256], r2[256];
  r1[tid] = s; r2[tid] = ss; __syncthreads();
  for (int st = 128; st > 0; st >>= 1) {
    if (tid < st) { r1[tid] += r1[tid + st]; r2[tid] += r2[tid + st]; }
    __syncthreads();
  }
  const float mean = r1[0] * (1.0f / DM);
  const float var  = r2[0] * (1.0f / DM) - mean * mean;
  const float rstd = rsqrtf(var + LN_EPS);
  v4h o;
#pragma unroll
  for (int j = 0; j < 4; ++j) {
    const int i = tid * 4 + j;
    o[j] = (_Float16)(((v[j] - mean) * rstd) * gamma[i] + beta[i]);
  }
  *reinterpret_cast<v4h*>(xh + (size_t)row * DM + tid * 4) = o;
}

// ---------------- 2) f32 -> f16 convert ----------------
__global__ void cvt_kernel(const float* __restrict__ in, _Float16* __restrict__ out, int n) {
  const int i = blockIdx.x * 256 + threadIdx.x;
  if (i < n) out[i] = (_Float16)in[i];
}

// ---------------- 3) GEMM: C[M,N] = A[M,K] * Bw[N,K]^T + bias, *scale ----------------
// block = 128 threads (4 waves). Block tile = 32(M) x 256(N); wave w owns the
// 64-wide N slice w. A tile (32x32 halves) is staged in LDS (async copy,
// double-buffered) and shared by all 4 waves; B streams from global (L2-hot).
__global__ void gemm_f16(const _Float16* __restrict__ A, const _Float16* __restrict__ Bw,
                         const float* __restrict__ bias, float* __restrict__ C,
                         int N, int Kd, float scale) {
  const int lane = threadIdx.x & 31;
  const int wid  = __builtin_amdgcn_readfirstlane(threadIdx.x >> 5);
  const int hi   = lane >> 4;
  const int rl   = lane & 15;
  const int nbase = blockIdx.x * 256 + wid * 64;
  const int mbase = blockIdx.y * 32;

  __shared__ _Float16 sA[2][32 * LDS_STRIDE];

  // cooperative A-tile copy assignment: thread i copies 16B (row i>>2, seg i&3)
  const int crow = threadIdx.x >> 2;
  const int cseg = threadIdx.x & 3;
  const _Float16* csrc = A + (size_t)(mbase + crow) * Kd + cseg * 8;
  _Float16* cdst[2] = { &sA[0][crow * LDS_STRIDE + cseg * 8],
                        &sA[1][crow * LDS_STRIDE + cseg * 8] };

  const _Float16* brow0 = Bw + (size_t)(nbase +  0 + rl) * Kd;
  const _Float16* brow1 = Bw + (size_t)(nbase + 16 + rl) * Kd;
  const _Float16* brow2 = Bw + (size_t)(nbase + 32 + rl) * Kd;
  const _Float16* brow3 = Bw + (size_t)(nbase + 48 + rl) * Kd;

  v8f acc[2][4] = {};

  // prologue: stage k0 = 0 into buffer 0
#if defined(HAVE_ASYNC_LDS)
  __builtin_amdgcn_global_load_async_to_lds_b128((as1_v4i*)csrc, (as3_v4i*)cdst[0], 0, 0);
#else
  *reinterpret_cast<v8h*>(cdst[0]) = *reinterpret_cast<const v8h*>(csrc);
#endif
  wait_async_copies();
  __syncthreads();

  int buf = 0;
  for (int k0 = 0; k0 < Kd; k0 += 32) {
    if (k0 + 32 < Kd) {
#if defined(HAVE_ASYNC_LDS)
      __builtin_amdgcn_global_load_async_to_lds_b128(
          (as1_v4i*)(csrc + k0 + 32), (as3_v4i*)cdst[buf ^ 1], 0, 0);
#else
      *reinterpret_cast<v8h*>(cdst[buf ^ 1]) =
          *reinterpret_cast<const v8h*>(csrc + k0 + 32);
#endif
    }
    __builtin_prefetch(brow0 + k0 + 512, 0, 3);
    __builtin_prefetch(brow3 + k0 + 512, 0, 3);

    const _Float16* tile = &sA[buf][0];
    v16h a0 = lds_op16(tile, rl, hi);
    v16h a1 = lds_op16(tile, rl + 16, hi);
    v16h b0 = load_op16(brow0, k0, hi);
    v16h b1 = load_op16(brow1, k0, hi);
    v16h b2 = load_op16(brow2, k0, hi);
    v16h b3 = load_op16(brow3, k0, hi);

    acc[0][0] = __builtin_amdgcn_wmma_f32_16x16x32_f16(false, a0, false, b0, (short)0, acc[0][0], false, false);
    acc[0][1] = __builtin_amdgcn_wmma_f32_16x16x32_f16(false, a0, false, b1, (short)0, acc[0][1], false, false);
    acc[0][2] = __builtin_amdgcn_wmma_f32_16x16x32_f16(false, a0, false, b2, (short)0, acc[0][2], false, false);
    acc[0][3] = __builtin_amdgcn_wmma_f32_16x16x32_f16(false, a0, false, b3, (short)0, acc[0][3], false, false);
    acc[1][0] = __builtin_amdgcn_wmma_f32_16x16x32_f16(false, a1, false, b0, (short)0, acc[1][0], false, false);
    acc[1][1] = __builtin_amdgcn_wmma_f32_16x16x32_f16(false, a1, false, b1, (short)0, acc[1][1], false, false);
    acc[1][2] = __builtin_amdgcn_wmma_f32_16x16x32_f16(false, a1, false, b2, (short)0, acc[1][2], false, false);
    acc[1][3] = __builtin_amdgcn_wmma_f32_16x16x32_f16(false, a1, false, b3, (short)0, acc[1][3], false, false);

    wait_async_copies();
    __syncthreads();
    buf ^= 1;
  }

  float bn[4];
#pragma unroll
  for (int ni = 0; ni < 4; ++ni) bn[ni] = bias[nbase + ni * 16 + rl];
#pragma unroll
  for (int mi = 0; mi < 2; ++mi) {
#pragma unroll
    for (int r = 0; r < 8; ++r) {
      const int m = mbase + mi * 16 + r + 8 * hi;
      float* cp = C + (size_t)m * N + nbase + rl;
#pragma unroll
      for (int ni = 0; ni < 4; ++ni)
        cp[ni * 16] = (acc[mi][ni][r] + bn[ni]) * scale;
    }
  }
}

// ---------------- 4) L2-norm Q,K; split into Qh,Kh [B,H,T,DK] and Vt [B,H,DK,T] ----------------
__global__ void normsplit_kernel(const float* __restrict__ qkv, _Float16* __restrict__ Qh,
                                 _Float16* __restrict__ Kh, _Float16* __restrict__ Vt) {
  const int h = blockIdx.x;               // 0..15
  const int trow = blockIdx.y;            // 0..BT-1
  const int d = threadIdx.x;              // 0..63
  const int b = trow / TT, t = trow % TT;
  const float* rp = qkv + (size_t)trow * N3;
  const float qv = rp[h * DK + d];
  const float kv = rp[DM + h * DK + d];
  const float vv = rp[2 * DM + h * DK + d];
  __shared__ float sq[64], sk[64];
  sq[d] = qv * qv; sk[d] = kv * kv; __syncthreads();
  for (int st = 32; st > 0; st >>= 1) {
    if (d < st) { sq[d] += sq[d + st]; sk[d] += sk[d + st]; }
    __syncthreads();
  }
  const float rq = 1.0f / fmaxf(sqrtf(sq[0]), NORM_EPS);
  const float rk = 1.0f / fmaxf(sqrtf(sk[0]), NORM_EPS);
  const size_t bh = (size_t)b * NH + h;
  const size_t idx = (bh * TT + t) * DK + d;
  Qh[idx] = (_Float16)(qv * rq);
  Kh[idx] = (_Float16)(kv * rk);
  Vt[(bh * DK + d) * TT + t] = (_Float16)vv;
}

// ---------------- 5) scores = clip(Q K^T * SCALE) + masks -> attn (raw) ----------------
// grid: (T/256, T/16, B*H), block 128. Wave w covers a 64-wide K strip; the A
// (Q) operand is loaded once and reused for 4 K-subtiles. All causal-skip
// branches are scalar (readfirstlane) so EXEC stays all-ones around WMMA.
__global__ void scores_kernel(const _Float16* __restrict__ Qh, const _Float16* __restrict__ Kh,
                              const unsigned char* __restrict__ pad, float* __restrict__ attn) {
  const int lane = threadIdx.x & 31;
  const int wid  = __builtin_amdgcn_readfirstlane(threadIdx.x >> 5);
  const int hi   = lane >> 4;
  const int rl   = lane & 15;
  const int bh    = blockIdx.z;
  const int b     = bh >> 4;
  const int qbase = blockIdx.y * 16;
  const int kbase = blockIdx.x * 256 + wid * 64;
  float* out = attn + (size_t)bh * TT * TT;

  const int nlive = min(4, max(0, (qbase + 16 - kbase + 15) >> 4));  // scalar
  v8f acc[4] = {};
  if (nlive > 0) {
    const _Float16* arow = Qh + ((size_t)bh * TT + qbase + rl) * DK;
#pragma unroll
    for (int k0 = 0; k0 < DK; k0 += 32) {
      v16h a = load_op16(arow, k0, hi);
#pragma unroll
      for (int kt = 0; kt < 4; ++kt) {
        if (kt < nlive) {
          const _Float16* brow = Kh + ((size_t)bh * TT + kbase + kt * 16 + rl) * DK;
          v16h bm = load_op16(brow, k0, hi);
          acc[kt] = __builtin_amdgcn_wmma_f32_16x16x32_f16(false, a, false, bm, (short)0, acc[kt], false, false);
        }
      }
    }
  }
#pragma unroll
  for (int kt = 0; kt < 4; ++kt) {
    const int kg = kbase + kt * 16 + rl;
    const bool pm = pad[(size_t)b * TT + kg] != 0;
#pragma unroll
    for (int r = 0; r < 8; ++r) {
      const int q = qbase + r + 8 * hi;
      float s = NEGV;
      if (kt < nlive) {
        s = fminf(10.0f, fmaxf(-10.0f, acc[kt][r] * SCALE));
        if (kg > q || pm) s = NEGV;
      }
      out[(size_t)q * TT + kg] = s;
    }
  }
}

// ---------------- 6) row softmax over attn (T=2048, 256 thr x 8) ----------------
__global__ void softmax_kernel(float* __restrict__ attn) {
  const size_t row = blockIdx.x;
  float* p = attn + row * (size_t)TT;
  const int tid = threadIdx.x;
  v4f va = *reinterpret_cast<const v4f*>(p + tid * 8);
  v4f vb = *reinterpret_cast<const v4f*>(p + tid * 8 + 4);
  float m = -3.4e38f;
#pragma unroll
  for (int j = 0; j < 4; ++j) { m = fmaxf(m, va[j]); m = fmaxf(m, vb[j]); }
  __shared__ float red[256];
  red[tid] = m; __syncthreads();
  for (int st = 128; st > 0; st >>= 1) {
    if (tid < st) red[tid] = fmaxf(red[tid], red[tid + st]);
    __syncthreads();
  }
  const float rm = red[0]; __syncthreads();
  float s = 0.0f;
#pragma unroll
  for (int j = 0; j < 4; ++j) {
    va[j] = __expf(va[j] - rm); s += va[j];
    vb[j] = __expf(vb[j] - rm); s += vb[j];
  }
  red[tid] = s; __syncthreads();
  for (int st = 128; st > 0; st >>= 1) {
    if (tid < st) red[tid] += red[tid + st];
    __syncthreads();
  }
  const float inv = 1.0f / red[0];
#pragma unroll
  for (int j = 0; j < 4; ++j) { va[j] *= inv; vb[j] *= inv; }
  *reinterpret_cast<v4f*>(p + tid * 8) = va;
  *reinterpret_cast<v4f*>(p + tid * 8 + 4) = vb;
}

// ---------------- 7) attn @ V : Oh[b*T+q, h*64+d] (f16) ----------------
// grid: (T/64, B*H), block 128. Wave w owns q-tile w; each wave computes
// 16(q) x 64(d): the attn (A) operand is loaded/converted once per K-step and
// reused across 4 d-subtiles. K loop truncated at q+16 (attn==0 beyond, exact
// f32 underflow of exp(-1e4 - max)).
__global__ void av_kernel(const float* __restrict__ attn, const _Float16* __restrict__ Vt,
                          _Float16* __restrict__ Oh) {
  const int lane = threadIdx.x & 31;
  const int wid  = __builtin_amdgcn_readfirstlane(threadIdx.x >> 5);
  const int hi   = lane >> 4;
  const int rl   = lane & 15;
  const int bh    = blockIdx.y;
  const int qbase = blockIdx.x * 64 + wid * 16;
  const float*    arow = attn + ((size_t)bh * TT + qbase + rl) * TT;
  const _Float16* br0  = Vt + ((size_t)bh * DK +  0 + rl) * TT;
  const _Float16* br1  = Vt + ((size_t)bh * DK + 16 + rl) * TT;
  const _Float16* br2  = Vt + ((size_t)bh * DK + 32 + rl) * TT;
  const _Float16* br3  = Vt + ((size_t)bh * DK + 48 + rl) * TT;
  v8f acc[4] = {};
  const int kend = qbase + 16;            // scalar loop bound
  for (int k0 = 0; k0 < kend; k0 += 32) {
    const int o0 = k0 + (hi ? 8 : 0);
    const int o1 = k0 + (hi ? 24 : 16);
    v4f f0 = *reinterpret_cast<const v4f*>(arow + o0);
    v4f f1 = *reinterpret_cast<const v4f*>(arow + o0 + 4);
    v4f f2 = *reinterpret_cast<const v4f*>(arow + o1);
    v4f f3 = *reinterpret_cast<const v4f*>(arow + o1 + 4);
    v16h a;
#pragma unroll
    for (int i = 0; i < 4; ++i) {
      a[i]      = (_Float16)f0[i];
      a[4 + i]  = (_Float16)f1[i];
      a[8 + i]  = (_Float16)f2[i];
      a[12 + i] = (_Float16)f3[i];
    }
    v16h b0 = load_op16(br0, k0, hi);
    v16h b1 = load_op16(br1, k0, hi);
    v16h b2 = load_op16(br2, k0, hi);
    v16h b3 = load_op16(br3, k0, hi);
    acc[0] = __builtin_amdgcn_wmma_f32_16x16x32_f16(false, a, false, b0, (short)0, acc[0], false, false);
    acc[1] = __builtin_amdgcn_wmma_f32_16x16x32_f16(false, a, false, b1, (short)0, acc[1], false, false);
    acc[2] = __builtin_amdgcn_wmma_f32_16x16x32_f16(false, a, false, b2, (short)0, acc[2], false, false);
    acc[3] = __builtin_amdgcn_wmma_f32_16x16x32_f16(false, a, false, b3, (short)0, acc[3], false, false);
  }
  const int b = bh >> 4, h = bh & 15;
#pragma unroll
  for (int r = 0; r < 8; ++r) {
    const int q = qbase + r + 8 * hi;
    _Float16* op = Oh + ((size_t)b * TT + q) * DM + h * DK + rl;
#pragma unroll
    for (int ni = 0; ni < 4; ++ni) op[ni * 16] = (_Float16)acc[ni][r];
  }
}

extern "C" void kernel_launch(void* const* d_in, const int* in_sizes, int n_in,
                              void* d_out, int out_size, void* d_ws, size_t ws_size,
                              hipStream_t stream) {
  (void)in_sizes; (void)n_in; (void)out_size; (void)ws_size;
  const float* x      = (const float*)d_in[0];
  const float* lng    = (const float*)d_in[1];
  const float* lnb    = (const float*)d_in[2];
  const float* qkv_w  = (const float*)d_in[3];
  const float* qkv_b  = (const float*)d_in[4];
  const float* out_w  = (const float*)d_in[5];
  const float* out_b  = (const float*)d_in[6];
  const unsigned char* pad = (const unsigned char*)d_in[7];

  float* outp = (float*)d_out;                        // [B,T,D]
  float* attn = outp + (size_t)BT * DM;               // [B,H,T,T]

  char* w = (char*)d_ws;
  _Float16* xh     = (_Float16*)w;                 w += (size_t)BT * DM * 2;       // 8 MB
  _Float16* wqkvh  = (_Float16*)w;                 w += (size_t)N3 * DM * 2;       // 6 MB
  _Float16* wouth  = (_Float16*)w;                 w += (size_t)DM * DM * 2;       // 2 MB
  float*    qkvf   = (float*)w;                    w += (size_t)BT * N3 * 4;       // 50 MB
  _Float16* Qh     = (_Float16*)w;                 w += (size_t)BB * NH * TT * DK * 2;
  _Float16* Kh     = (_Float16*)w;                 w += (size_t)BB * NH * TT * DK * 2;
  _Float16* Vt     = (_Float16*)w;                 w += (size_t)BB * NH * DK * TT * 2;
  _Float16* Oh     = (_Float16*)w;                 w += (size_t)BT * DM * 2;       // total ~100 MB

  // 1) LayerNorm -> f16
  ln_kernel<<<BT, 256, 0, stream>>>(x, lng, lnb, xh);
  // 2) weights -> f16
  cvt_kernel<<<(N3 * DM + 255) / 256, 256, 0, stream>>>(qkv_w, wqkvh, N3 * DM);
  cvt_kernel<<<(DM * DM + 255) / 256, 256, 0, stream>>>(out_w, wouth, DM * DM);
  // 3) QKV projection: [BT,3072] = xh * wqkvh^T + qkv_b
  gemm_f16<<<dim3(N3 / 256, BT / 32), 128, 0, stream>>>(xh, wqkvh, qkv_b, qkvf, N3, DM, 1.0f);
  // 4) l2norm + split to Qh/Kh/Vt
  normsplit_kernel<<<dim3(NH, BT), 64, 0, stream>>>(qkvf, Qh, Kh, Vt);
  // 5) masked clipped scores -> attn region of d_out
  scores_kernel<<<dim3(TT / 256, TT / 16, BB * NH), 128, 0, stream>>>(Qh, Kh, pad, attn);
  // 6) softmax in place
  softmax_kernel<<<BB * NH * TT, 256, 0, stream>>>(attn);
  // 7) attn @ V -> Oh (f16, [B,T,D] with heads merged)
  av_kernel<<<dim3(TT / 64, BB * NH), 128, 0, stream>>>(attn, Vt, Oh);
  // 8) out projection (+bias) * 0.5 -> d_out
  gemm_f16<<<dim3(DM / 256, BT / 32), 128, 0, stream>>>(Oh, wouth, out_b, outp, DM, DM, 0.5f);
}
